// SimpleRNN7_22660247454199
// MI455X (gfx1250) — matrix operand order repaired and checked by reading, not compiled, not run on hardware
//
#include <hip/hip_runtime.h>

// ---------------------------------------------------------------------------
// GRU-gated neural-ODE scan for MI455X (gfx1250, wave32, WMMA).
// One workgroup (8 waves, 256 threads) owns a 16-sample batch tile and runs
// the full K=1024 recurrence with weights resident in LDS (bf16) and the GRU
// hidden state resident in WMMA accumulator fragments.
// ---------------------------------------------------------------------------

typedef __attribute__((ext_vector_type(16))) __bf16 v16bf;
typedef __attribute__((ext_vector_type(8)))  float  v8f;

#define B_TOT   256
#define K_STEPS 1024
#define HID     128
#define G3      384      // 3*HID gate rows
#define LIFT    32
#define MT      16       // batch tile per block (== WMMA M)

__device__ __forceinline__ float sigf(float x) { return 1.0f / (1.0f + __expf(-x)); }

// A-matrix fragment (16x32 bf16, M = lane&15):
//   VGPR0-3 : K = kt*32 + 8*(lane>=16) + {0..7}
//   VGPR4-7 : K = kt*32 + 16 + 8*(lane>=16) + {0..7}
__device__ __forceinline__ v16bf load_a_frag(const __bf16* row, int kt, int lhalf) {
  int kb = kt * 32 + lhalf * 8;
  v16bf a;
  #pragma unroll
  for (int i = 0; i < 8; ++i) { a[i] = row[kb + i]; a[8 + i] = row[kb + 16 + i]; }
  return a;
}

// B-matrix fragment (32x16 bf16, N = lane&15):
//   VGPR0-3 : K = kt*32 + 8*(lane>=16) + {0..7}
//   VGPR4-7 : K = kt*32 + 16 + 8*(lane>=16) + {0..7}
__device__ __forceinline__ v16bf load_b_frag(const __bf16* row, int kt, int lhalf) {
  int kb = kt * 32 + lhalf * 8;
  v16bf b;
  #pragma unroll
  for (int i = 0; i < 8; ++i) { b[i] = row[kb + i]; b[8 + i] = row[kb + 16 + i]; }
  return b;
}

// 7-species kinetics RHS
__device__ __forceinline__ void rhs7(const float* y, const float* th, float* d) {
  float A = y[0], D = y[1], G = y[2], J = y[3], Kv = y[4], L = y[5], M = y[6];
  float kfAD = th[0], kfDG = th[1], kfGJ = th[2], kf10 = th[3], kf11 = th[4],
        kf12 = th[5], krAD = th[6], krDG = th[7], krGJ = th[8], kr11 = th[9],
        kr12 = th[10];
  d[0] = -kfAD * A + krAD * D;
  d[1] =  kfAD * A - krAD * D - kfDG * D + krDG * G;
  d[2] =  kfDG * D - krDG * G - kfGJ * G + krGJ * J;
  d[3] =  kfGJ * G - krGJ * J - kf10 * J;
  d[4] =  kf10 * J - kf11 * Kv + kr11 * L;
  d[5] =  kf11 * Kv - kr11 * L - kf12 * L + kr12 * M;
  d[6] =  kf12 * L - kr12 * M;
}

__global__ __launch_bounds__(256, 1)
void gruode_scan_kernel(const float* __restrict__ y0_obs,   // [256,3]
                        const float* __restrict__ u_seq,    // [256,1024,4]
                        const float* __restrict__ dt_seq,   // [256,1024]
                        const float* __restrict__ y_seq,    // [256,1024,3]
                        const float* __restrict__ lift_W,   // [32,7]
                        const float* __restrict__ lift_b,   // [32]
                        const float* __restrict__ W_ih,     // [384,32]
                        const float* __restrict__ W_hh,     // [384,128]
                        const float* __restrict__ b_ih,     // [384]
                        const float* __restrict__ b_hh,     // [384]
                        const float* __restrict__ head_W,   // [11,128]
                        const float* __restrict__ head_b,   // [11]
                        const float* __restrict__ u2x,      // [4,7]
                        float* __restrict__ out_y,          // [256,1024,3]
                        float* __restrict__ out_th)         // [256,1024,11]
{
  // big weight tiles in dynamic LDS (bf16): W_hh 96KB + W_ih 24KB + head 4KB
  extern __shared__ __bf16 dynlds[];
  __bf16* Whh_s = dynlds;                    // [384][128]
  __bf16* Wih_s = Whh_s + G3 * HID;          // [384][32]
  __bf16* Hdw_s = Wih_s + G3 * LIFT;         // [16][128] (rows 11..15 zero)

  __shared__ __bf16 h_bf[MT * HID];          // hidden state broadcast (bf16)
  __shared__ __bf16 z_bf[MT * LIFT];         // lift activations (bf16)
  __shared__ float  feat_s[MT * 8];          // [u0..u3, xobs0..2, pad]
  __shared__ float  x_state[MT * 8];         // ODE state (7 species + pad)
  __shared__ float  theta_s[MT * 12];        // rates (11 + pad)
  __shared__ float  liftW_s[LIFT * 8];       // [32][8] padded
  __shared__ float  liftb_s[LIFT];
  __shared__ float  headb_s[16];
  __shared__ float  u2x_s[4 * 8];            // [4][8] padded

  const int t     = threadIdx.x;
  const int wave  = t >> 5;
  const int lane  = t & 31;
  const int lmod  = lane & 15;
  const int lhalf = lane >> 4;
  const int b0    = blockIdx.x * MT;

  // ---------------- one-time weight staging (f32 -> bf16) ----------------
  for (int i = t; i < G3 * HID; i += 256)  Whh_s[i] = (__bf16)W_hh[i];
  for (int i = t; i < G3 * LIFT; i += 256) Wih_s[i] = (__bf16)W_ih[i];
  for (int i = t; i < 16 * HID; i += 256) {
    int n = i >> 7, kk = i & 127;
    Hdw_s[i] = (n < 11) ? (__bf16)head_W[n * HID + kk] : (__bf16)0.0f;
  }
  for (int i = t; i < LIFT * 8; i += 256) {
    int o = i >> 3, f = i & 7;
    liftW_s[i] = (f < 7) ? lift_W[o * 7 + f] : 0.0f;
  }
  if (t < LIFT) liftb_s[t] = lift_b[t];
  if (t < 16)   headb_s[t] = (t < 11) ? head_b[t] : 0.0f;
  if (t < 32) { int j = t >> 3, s = t & 7; u2x_s[t] = (s < 7) ? u2x[j * 7 + s] : 0.0f; }
  if (t < MT * 8) {
    int m = t >> 3, s = t & 7;
    float v = 0.01f;                        // x0 = set_obs(y0) + 0.01 everywhere
    if (s == 0) v += y0_obs[(b0 + m) * 3 + 0];
    if (s == 3) v += y0_obs[(b0 + m) * 3 + 1];
    if (s == 6) v += y0_obs[(b0 + m) * 3 + 2];
    x_state[t] = (s < 7) ? v : 0.0f;
  }
  for (int i = t; i < MT * HID; i += 256) h_bf[i] = (__bf16)0.0f;   // h0 = 0
  __syncthreads();

  // ------------- per-wave persistent B-fragments (weights) ---------------
  // wave w owns gate N-tiles {w, w+8, w+16}: i_r / i_z / i_n columns 16w..16w+15
  const int ncol = 16 * wave + lmod;
  v16bf Bih[3], Bhh[3][4], Bhd[4];
  float bias_i[3], bias_h[3];
  #pragma unroll
  for (int g = 0; g < 3; ++g) {
    int n = ncol + g * HID;
    Bih[g] = load_b_frag(Wih_s + n * LIFT, 0, lhalf);
    #pragma unroll
    for (int kt = 0; kt < 4; ++kt)
      Bhh[g][kt] = load_b_frag(Whh_s + n * HID, kt, lhalf);
    bias_i[g] = b_ih[n];
    bias_h[g] = b_hh[n];
  }
  #pragma unroll
  for (int kt = 0; kt < 4; ++kt)
    Bhd[kt] = load_b_frag(Hdw_s + lmod * HID, kt, lhalf);

  float hreg[8];                             // h slice [16 samples][16 cols] frag
  #pragma unroll
  for (int r = 0; r < 8; ++r) hreg[r] = 0.0f;

  // ------------------------------ scan ------------------------------------
  for (int k = 0; k < K_STEPS; ++k) {
    // Phase A1: features (u_k + observed state, with teacher forcing)
    if (t < MT) {
      const int m = t, b = b0 + m;
      const float* u = u_seq + ((size_t)b * K_STEPS + k) * 4;
      feat_s[m * 8 + 0] = u[0]; feat_s[m * 8 + 1] = u[1];
      feat_s[m * 8 + 2] = u[2]; feat_s[m * 8 + 3] = u[3];
      const bool tf = (k > 0) && (k % 50 == 0);
      const int obs[3] = {0, 3, 6};
      #pragma unroll
      for (int c = 0; c < 3; ++c) {
        float xo = tf ? y_seq[((size_t)b * K_STEPS + (k - 1)) * 3 + c]
                      : x_state[m * 8 + obs[c]];
        feat_s[m * 8 + 4 + c] = xo;
      }
      feat_s[m * 8 + 7] = 0.0f;
    }
    __syncthreads();

    // Phase A2: lift MLP (tiny, VALU): z = silu(feat @ lift_W^T + b)
    for (int jid = t; jid < MT * LIFT; jid += 256) {
      int o = jid & 31, m = jid >> 5;
      float a = liftb_s[o];
      #pragma unroll
      for (int f = 0; f < 7; ++f) a += feat_s[m * 8 + f] * liftW_s[o * 8 + f];
      z_bf[m * LIFT + o] = (__bf16)(a * sigf(a));
    }
    __syncthreads();

    // Phase B: GRU GEMMs via v_wmma_f32_16x16x32_bf16 + gate math in fragments
    v8f acc_i[3], acc_h[3];
    #pragma unroll
    for (int g = 0; g < 3; ++g)
      #pragma unroll
      for (int r = 0; r < 8; ++r) { acc_i[g][r] = bias_i[g]; acc_h[g][r] = bias_h[g]; }

    {
      v16bf Az = load_a_frag(z_bf + lmod * LIFT, 0, lhalf);
      #pragma unroll
      for (int g = 0; g < 3; ++g)
        acc_i[g] = __builtin_amdgcn_wmma_f32_16x16x32_bf16(
            false, Az, false, Bih[g], (short)0, acc_i[g], false, false);
      #pragma unroll
      for (int kt = 0; kt < 4; ++kt) {
        v16bf Ah = load_a_frag(h_bf + lmod * HID, kt, lhalf);
        #pragma unroll
        for (int g = 0; g < 3; ++g)
          acc_h[g] = __builtin_amdgcn_wmma_f32_16x16x32_bf16(
              false, Ah, false, Bhh[g][kt], (short)0, acc_h[g], false, false);
      }
    }
    #pragma unroll
    for (int r = 0; r < 8; ++r) {
      float rg = sigf(acc_i[0][r] + acc_h[0][r]);
      float zg = sigf(acc_i[1][r] + acc_h[1][r]);
      float ng = tanhf(acc_i[2][r] + rg * acc_h[2][r]);
      hreg[r]  = (1.0f - zg) * ng + zg * hreg[r];
    }
    __syncthreads();                         // everyone done reading old h_bf
    #pragma unroll
    for (int r = 0; r < 8; ++r)
      h_bf[(r + 8 * lhalf) * HID + 16 * wave + lmod] = (__bf16)hreg[r];
    __syncthreads();                         // h_new visible to all waves

    // Phase C: head GEMM (M=16,N=16(pad11),K=128) on wave 0 -> theta
    if (wave == 0) {
      v8f acc;
      #pragma unroll
      for (int r = 0; r < 8; ++r) acc[r] = 0.0f;
      #pragma unroll
      for (int kt = 0; kt < 4; ++kt) {
        v16bf Ah = load_a_frag(h_bf + lmod * HID, kt, lhalf);
        acc = __builtin_amdgcn_wmma_f32_16x16x32_bf16(
            false, Ah, false, Bhd[kt], (short)0, acc, false, false);
      }
      float hb = headb_s[lmod];
      #pragma unroll
      for (int r = 0; r < 8; ++r) {
        float th = 0.01f + 2.99f * sigf(acc[r] + hb);   // _gamma
        int m = r + 8 * lhalf;
        if (lmod < 11) {
          theta_s[m * 12 + lmod] = th;
          out_th[((size_t)(b0 + m) * K_STEPS + k) * 11 + lmod] = th;
        }
      }
    }
    __syncthreads();

    // Phase D: per-sample input jump + 10 RK4 substeps (scalar, lanes 0..15)
    if (t < MT) {
      const int m = t, b = b0 + m;
      float th[11];
      #pragma unroll
      for (int i = 0; i < 11; ++i) th[i] = theta_s[m * 12 + i];
      float x[7];
      #pragma unroll
      for (int s = 0; s < 7; ++s) {
        float a = x_state[m * 8 + s];
        #pragma unroll
        for (int j = 0; j < 4; ++j) a += feat_s[m * 8 + j] * u2x_s[j * 8 + s];
        x[s] = a;                            // x_jump
      }
      const float hsub = dt_seq[(size_t)b * K_STEPS + k] * 0.1f;
      for (int sub = 0; sub < 10; ++sub) {
        float k1[7], k2[7], k3[7], k4[7], tmp[7];
        rhs7(x, th, k1);
        #pragma unroll
        for (int s = 0; s < 7; ++s) tmp[s] = x[s] + 0.5f * hsub * k1[s];
        rhs7(tmp, th, k2);
        #pragma unroll
        for (int s = 0; s < 7; ++s) tmp[s] = x[s] + 0.5f * hsub * k2[s];
        rhs7(tmp, th, k3);
        #pragma unroll
        for (int s = 0; s < 7; ++s) tmp[s] = x[s] + hsub * k3[s];
        rhs7(tmp, th, k4);
        #pragma unroll
        for (int s = 0; s < 7; ++s)
          x[s] = fmaxf(x[s] + hsub * (1.0f / 6.0f) *
                                  (k1[s] + 2.0f * k2[s] + 2.0f * k3[s] + k4[s]),
                       0.0f);
      }
      #pragma unroll
      for (int s = 0; s < 7; ++s) x_state[m * 8 + s] = x[s];
      float* oy = out_y + ((size_t)b * K_STEPS + k) * 3;
      oy[0] = x[0]; oy[1] = x[3]; oy[2] = x[6];
    }
    __syncthreads();
  }
}

extern "C" void kernel_launch(void* const* d_in, const int* in_sizes, int n_in,
                              void* d_out, int out_size, void* d_ws, size_t ws_size,
                              hipStream_t stream) {
  (void)in_sizes; (void)n_in; (void)d_ws; (void)ws_size; (void)out_size;
  const float* y0_obs = (const float*)d_in[0];
  const float* u_seq  = (const float*)d_in[1];
  const float* dt_seq = (const float*)d_in[2];
  const float* y_seq  = (const float*)d_in[3];
  const float* lift_W = (const float*)d_in[4];
  const float* lift_b = (const float*)d_in[5];
  const float* W_ih   = (const float*)d_in[6];
  const float* W_hh   = (const float*)d_in[7];
  const float* b_ih   = (const float*)d_in[8];
  const float* b_hh   = (const float*)d_in[9];
  const float* head_W = (const float*)d_in[10];
  const float* head_b = (const float*)d_in[11];
  const float* u2x    = (const float*)d_in[12];

  float* out_y  = (float*)d_out;
  float* out_th = out_y + (size_t)B_TOT * K_STEPS * 3;

  const size_t dyn_bytes = (size_t)(G3 * HID + G3 * LIFT + 16 * HID) * sizeof(__bf16);
  gruode_scan_kernel<<<dim3(B_TOT / MT), dim3(256), dyn_bytes, stream>>>(
      y0_obs, u_seq, dt_seq, y_seq, lift_W, lift_b, W_ih, W_hh, b_ih, b_hh,
      head_W, head_b, u2x, out_y, out_th);
}
